// SimpleEquivariantGNN_19550691132075
// MI455X (gfx1250) — matrix operand order repaired and checked by reading, not compiled, not run on hardware
//
#include <hip/hip_runtime.h>

// SimpleEquivariantGNN fused head for MI455X (gfx1250, wave32, WMMA).
// Per 128-atom tile (one workgroup, 8 waves):
//   gather t_embedding[batch] -> LDS f16 (overlapped with async weight stage)
//   weights staged once per block: global_load_async_to_lds_b128 (ASYNCcnt)
//   h  = silu(h0 @ W1 + b1) @ W2 + b2            (v_wmma_f32_16x16x32_f16)
//   logits = h @ Wt + bt  -> d_out               (WMMA, 112-col padded)
//   u  = silu(h @ Wc1 + bc1)                     (WMMA)
//   coord = u @ wc2 + bc2 -> d_out               (scalar, K=128)
//   h_crystal[b] += h      (run-length + global_atomic_add_f32)
// Then cell_head: cell_update = silu(h_crystal @ Wl1 + bl1) @ Wl2 + bl2.
// WMMA K-loop is software-pipelined (next fragments prefetched into fresh
// registers) and processes two N-tiles per A fragment: no WAR v_nops, partial
// dscnt waits, half the A-side LDS traffic.

#define NN 500000
#define BB 4096
#define HH 256
#define TM 128          // atom rows per workgroup tile
#define RS 264          // LDS f16 row stride for Ah/Bh (256 + 8 pad)
#define WBS 264         // LDS f16 row stride for staged weights
#define CB 64           // crystals per block in cell_head

// LDS byte offsets inside dynamic smem (fused kernel)
#define AH_OFF  0
#define BH_OFF  (TM * RS * 2)                    // 67584
#define WB_OFF  (2 * TM * RS * 2)                // 135168
#define SB_OFF  (WB_OFF + HH * WBS * 2)          // 270336
#define WC2_OFF (SB_OFF + TM * 4)                // 270848
#define SMEM_FUSED (WC2_OFF + 388 * 4)           // 272400 B (< 320 KB/WG)

typedef _Float16 f16;
typedef __attribute__((ext_vector_type(16))) _Float16 v16h;
typedef __attribute__((ext_vector_type(8)))  _Float16 v8h;
typedef __attribute__((ext_vector_type(8)))  float    v8f;

struct alignas(8) H4 { f16 x, y, z, w; };

__device__ __forceinline__ float silu_f(float x) {
  return x / (1.0f + __expf(-x));
}

// Issue async DMA of ncols rows of Wg ([Nout][256] f16, contiguous) into the
// LDS weight buffer (stride WBS f16). 16B chunk per lane per op; ASYNCcnt.
__device__ __forceinline__ void stage_w_issue(const f16* Wg, const f16* Wb,
                                              int ncols, int tid) {
  const uint32_t base = (uint32_t)(uintptr_t)Wb;  // wave-relative LDS offset
  const int chunks = ncols * 32;                  // 32 x 16B chunks per column
  for (int i = tid; i < chunks; i += 256) {
    const int col = i >> 5;
    const int kc  = i & 31;
    const uint32_t lds = base + (uint32_t)(col * (WBS * 2) + kc * 16);
    const unsigned long long g =
        (unsigned long long)(uintptr_t)((const char*)(Wg + (size_t)col * HH) + kc * 16);
    asm volatile("global_load_async_to_lds_b128 %0, %1, off"
                 :: "v"(lds), "v"(g) : "memory");
  }
}

__device__ __forceinline__ void stage_w_wait() {
  asm volatile("s_wait_asynccnt 0x0" ::: "memory");
}

// A fragment (16x32 f16, ISA 7.12.2): lanes 0-15 row=lane, K {k0+0..7, k0+16..23};
// lanes 16-31 row=lane-16, K {k0+8..15, k0+24..31}. Two ds_load_b128 per lane.
__device__ __forceinline__ v16h ld_afrag(const f16* A, int m0, int k0, int lane) {
  const int half = lane >> 4;
  const int row  = m0 + (lane & 15);
  const f16* p = A + row * RS + k0 + half * 8;
  v8h lo = *(const v8h*)p;
  v8h hi = *(const v8h*)(p + 16);
  v16h r;
#pragma unroll
  for (int i = 0; i < 8; ++i) { r[i] = lo[i]; r[i + 8] = hi[i]; }
  return r;
}

// B fragment (32x16 f16) from the staged LDS weight buffer: lane holds column
// n0+(lane&15), K {ks+0..7, ks+16..23} with ks = k0 + 8*(lane>>4). Two ds_load_b128.
__device__ __forceinline__ v16h ld_bfrag(const f16* Wb, int n0, int k0, int lane) {
  const int col = n0 + (lane & 15);
  const int ks  = k0 + (lane >> 4) * 8;
  const f16* p = Wb + col * WBS + ks;
  v8h lo = *(const v8h*)p;
  v8h hi = *(const v8h*)(p + 16);
  v16h r;
#pragma unroll
  for (int i = 0; i < 8; ++i) { r[i] = lo[i]; r[i + 8] = hi[i]; }
  return r;
}

#define WMMA_F16(a, b, c) \
  __builtin_amdgcn_wmma_f32_16x16x32_f16(false, (a), false, (b), (short)0, (c), \
                                         false, false)

// Single 16x16 tile, K=256, 2-stage software pipeline.
__device__ __forceinline__ v8f wmma_k256(const f16* A, const f16* Wb,
                                         int m0, int n0, int lane) {
  v8f acc = {};
  v16h a = ld_afrag(A, m0, 0, lane);
  v16h b = ld_bfrag(Wb, n0, 0, lane);
#pragma unroll
  for (int kk = 32; kk < HH; kk += 32) {
    v16h an = ld_afrag(A, m0, kk, lane);
    v16h bn = ld_bfrag(Wb, n0, kk, lane);
    acc = WMMA_F16(a, b, acc);
    a = an; b = bn;
  }
  return WMMA_F16(a, b, acc);
}

// Two adjacent 16x16 N-tiles sharing one A fragment, K=256, pipelined.
// Two independent WMMAs back-to-back per step keep the matrix pipe busy.
__device__ __forceinline__ void wmma_k256_x2(const f16* A, const f16* Wb,
                                             int m0, int n0, int lane,
                                             v8f& acc0, v8f& acc1) {
  acc0 = (v8f){}; acc1 = (v8f){};
  v16h a  = ld_afrag(A, m0, 0, lane);
  v16h b0 = ld_bfrag(Wb, n0, 0, lane);
  v16h b1 = ld_bfrag(Wb, n0 + 16, 0, lane);
#pragma unroll
  for (int kk = 32; kk < HH; kk += 32) {
    v16h an  = ld_afrag(A, m0, kk, lane);
    v16h b0n = ld_bfrag(Wb, n0, kk, lane);
    v16h b1n = ld_bfrag(Wb, n0 + 16, kk, lane);
    acc0 = WMMA_F16(a, b0, acc0);
    acc1 = WMMA_F16(a, b1, acc1);
    a = an; b0 = b0n; b1 = b1n;
  }
  acc0 = WMMA_F16(a, b0, acc0);
  acc1 = WMMA_F16(a, b1, acc1);
}

// Transpose + f32->f16 convert: dst[n*K + k] = src[k*Nsrc + n] (zero-pad n>=Nsrc).
__global__ void prep_wT(const float* __restrict__ src, f16* __restrict__ dst,
                        int K, int Nsrc, int Ndst) {
  const int idx = blockIdx.x * 256 + threadIdx.x;
  if (idx >= Ndst * K) return;
  const int n = idx / K;
  const int k = idx - n * K;
  dst[idx] = (n < Nsrc) ? (f16)src[(size_t)k * Nsrc + n] : (f16)0.0f;
}

__global__ void fused_atoms(const int* __restrict__ batch,
                            const float* __restrict__ temb,
                            const f16* __restrict__ W1T, const float* __restrict__ b1,
                            const f16* __restrict__ W2T, const float* __restrict__ b2,
                            const f16* __restrict__ WC1T, const float* __restrict__ bc1,
                            const float* __restrict__ wc2, const float* __restrict__ bc2,
                            const f16* __restrict__ WTT, const float* __restrict__ bt,
                            float* __restrict__ hc,
                            float* __restrict__ outCoord,
                            float* __restrict__ outLogit) {
  extern __shared__ char smem[];
  f16*   Ah     = (f16*)(smem + AH_OFF);    // [128][RS]
  f16*   Bh     = (f16*)(smem + BH_OFF);    // [128][RS]
  f16*   Wb     = (f16*)(smem + WB_OFF);    // [256][WBS] staged weight panel
  int*   sBatch = (int*)(smem + SB_OFF);    // [128]
  float* sWc2   = (float*)(smem + WC2_OFF); // 384 + 3

  const int tid  = threadIdx.x;
  const int lane = tid & 31;
  const int wid  = tid >> 5;
  const int m0   = wid * 16;               // this wave's output row tile
  const int a0   = blockIdx.x * TM;
  const int rows = min(TM, NN - a0);

  // kick off W1 DMA into LDS; it overlaps the gather below
  stage_w_issue(W1T, Wb, 256, tid);

  if (tid < TM)  sBatch[tid] = (tid < rows) ? batch[a0 + tid] : -1;
  if (tid < 384) sWc2[tid] = wc2[tid];
  if (tid < 3)   sWc2[384 + tid] = bc2[tid];
  __syncthreads();

  // ---- gather: Ah[r][:] = (f16) t_embedding[batch[a0+r]][:]  (2 threads/row)
  {
    const int r  = tid >> 1;
    const int c0 = (tid & 1) * 128;
    f16* dst = Ah + r * RS + c0;
    if (r < rows) {
      const float* src = temb + (size_t)sBatch[r] * HH + c0;
#pragma unroll 4
      for (int i = 0; i < 128; i += 4) {
        const float4 f = *(const float4*)(src + i);
        H4 h = { (f16)f.x, (f16)f.y, (f16)f.z, (f16)f.w };
        *(H4*)(dst + i) = h;
      }
    } else {
      H4 z = {};
      for (int i = 0; i < 128; i += 4) *(H4*)(dst + i) = z;
    }
  }
  stage_w_wait();
  __syncthreads();

  // ---- GEMM1 + bias + SiLU -> Bh   (16 N-tiles as 8 pairs)
  for (int c = 0; c < 16; c += 2) {
    const int n0 = c * 16;
    v8f acc0, acc1;
    wmma_k256_x2(Ah, Wb, m0, n0, lane, acc0, acc1);
    const int ncol = n0 + (lane & 15);
    const float bv0 = b1[ncol];
    const float bv1 = b1[ncol + 16];
    const int mh = (lane >> 4) * 8;
#pragma unroll
    for (int i = 0; i < 8; ++i) {
      Bh[(m0 + mh + i) * RS + ncol]      = (f16)silu_f(acc0[i] + bv0);
      Bh[(m0 + mh + i) * RS + ncol + 16] = (f16)silu_f(acc1[i] + bv1);
    }
  }
  __syncthreads();

  stage_w_issue(W2T, Wb, 256, tid);
  stage_w_wait();
  __syncthreads();

  // ---- GEMM2 + bias -> Ah (this is h)
  for (int c = 0; c < 16; c += 2) {
    const int n0 = c * 16;
    v8f acc0, acc1;
    wmma_k256_x2(Bh, Wb, m0, n0, lane, acc0, acc1);
    const int ncol = n0 + (lane & 15);
    const float bv0 = b2[ncol];
    const float bv1 = b2[ncol + 16];
    const int mh = (lane >> 4) * 8;
#pragma unroll
    for (int i = 0; i < 8; ++i) {
      Ah[(m0 + mh + i) * RS + ncol]      = (f16)(acc0[i] + bv0);
      Ah[(m0 + mh + i) * RS + ncol + 16] = (f16)(acc1[i] + bv1);
    }
  }
  __syncthreads();

  stage_w_issue(WTT, Wb, 112, tid);
  stage_w_wait();
  __syncthreads();

  // ---- atom-type logits: h @ Wt + bt (112 cols: 3 tile-pairs + 1 single)
  {
    const int mh = (lane >> 4) * 8;
    for (int c = 0; c < 6; c += 2) {
      const int n0 = c * 16;
      v8f acc0, acc1;
      wmma_k256_x2(Ah, Wb, m0, n0, lane, acc0, acc1);
      const int ncol = n0 + (lane & 15);
      const float bv0 = (ncol < 100) ? bt[ncol] : 0.0f;
      const float bv1 = (ncol + 16 < 100) ? bt[ncol + 16] : 0.0f;
#pragma unroll
      for (int i = 0; i < 8; ++i) {
        const int ar = a0 + m0 + mh + i;
        if (ar < NN) {
          if (ncol < 100)      outLogit[(size_t)ar * 100 + ncol]      = acc0[i] + bv0;
          if (ncol + 16 < 100) outLogit[(size_t)ar * 100 + ncol + 16] = acc1[i] + bv1;
        }
      }
    }
    {
      const int n0 = 96;
      v8f acc = wmma_k256(Ah, Wb, m0, n0, lane);
      const int ncol = n0 + (lane & 15);
      const float bv = (ncol < 100) ? bt[ncol] : 0.0f;
#pragma unroll
      for (int i = 0; i < 8; ++i) {
        const int ar = a0 + m0 + mh + i;
        if (ar < NN && ncol < 100)
          outLogit[(size_t)ar * 100 + ncol] = acc[i] + bv;
      }
    }
  }
  __syncthreads();   // all waves done reading Wb before restaging

  stage_w_issue(WC1T, Wb, 128, tid);
  stage_w_wait();
  __syncthreads();

  // ---- coord hidden: u = silu(h @ Wc1 + bc1) -> Bh[:, 0:128]  (4 pairs)
  for (int c = 0; c < 8; c += 2) {
    const int n0 = c * 16;
    v8f acc0, acc1;
    wmma_k256_x2(Ah, Wb, m0, n0, lane, acc0, acc1);
    const int ncol = n0 + (lane & 15);
    const float bv0 = bc1[ncol];
    const float bv1 = bc1[ncol + 16];
    const int mh = (lane >> 4) * 8;
#pragma unroll
    for (int i = 0; i < 8; ++i) {
      Bh[(m0 + mh + i) * RS + ncol]      = (f16)silu_f(acc0[i] + bv0);
      Bh[(m0 + mh + i) * RS + ncol + 16] = (f16)silu_f(acc1[i] + bv1);
    }
  }
  __syncthreads();

  // ---- coord_update = u @ wc2 + bc2  (K=128, N=3 -> scalar, one thread/row)
  if (tid < rows) {
    const f16* u = Bh + tid * RS;
    float c0f = sWc2[384], c1f = sWc2[385], c2f = sWc2[386];
    for (int k = 0; k < 128; ++k) {
      const float uv = (float)u[k];
      c0f += uv * sWc2[k * 3 + 0];
      c1f += uv * sWc2[k * 3 + 1];
      c2f += uv * sWc2[k * 3 + 2];
    }
    float* o = outCoord + (size_t)(a0 + tid) * 3;
    o[0] = c0f; o[1] = c1f; o[2] = c2f;
  }

  // ---- segment-sum of h into h_crystal. batch is sorted: run-length locally,
  // flush with a global f32 atomic only when the crystal id changes.
  {
    const int col = tid;                   // 256 threads == 256 columns
    int curb = -1; float s = 0.0f;
    for (int r = 0; r < rows; ++r) {
      const int b = sBatch[r];
      if (b != curb) {
        if (curb >= 0) atomicAdd(&hc[(size_t)curb * HH + col], s);
        curb = b; s = 0.0f;
      }
      s += (float)Ah[r * RS + col];
    }
    if (curb >= 0) atomicAdd(&hc[(size_t)curb * HH + col], s);
  }
}

// cell_update = silu(h_crystal @ Wl1 + bl1) @ Wl2 + bl2, 64 crystals/block.
__global__ void cell_head(const float* __restrict__ hc,
                          const float* __restrict__ wl1, const float* __restrict__ bl1,
                          const float* __restrict__ wl2, const float* __restrict__ bl2,
                          float* __restrict__ outCell) {
  extern __shared__ char smem[];
  float* sW1 = (float*)smem;            // [256][128]
  float* sW2 = sW1 + 256 * 128;         // [128][6]
  float* sB1 = sW2 + 128 * 6;           // [128]
  float* sB2 = sB1 + 128;               // [6] (+2 pad)
  float* sH  = sB2 + 8;                 // [CB][256]
  float* sU  = sH + CB * 256;           // [CB][128]

  const int tid = threadIdx.x;
  const int c0  = blockIdx.x * CB;
  for (int i = tid; i < 256 * 128; i += 256) sW1[i] = wl1[i];
  for (int i = tid; i < 128 * 6;  i += 256)  sW2[i] = wl2[i];
  if (tid < 128) sB1[tid] = bl1[tid];
  if (tid < 6)   sB2[tid] = bl2[tid];
  for (int i = tid; i < CB * 256; i += 256)
    sH[i] = hc[(size_t)c0 * 256 + i];
  __syncthreads();

  for (int t = tid; t < CB * 128; t += 256) {
    const int cr = t >> 7, j = t & 127;
    float a = sB1[j];
    const float* hr = sH + cr * 256;
    for (int k = 0; k < 256; ++k) a += hr[k] * sW1[k * 128 + j];
    sU[cr * 128 + j] = a / (1.0f + __expf(-a));
  }
  __syncthreads();

  for (int t = tid; t < CB * 6; t += 256) {
    const int cr = t / 6, c = t - cr * 6;
    float a = sB2[c];
    const float* u = sU + cr * 128;
    for (int k = 0; k < 128; ++k) a += u[k] * sW2[k * 6 + c];
    outCell[(size_t)(c0 + cr) * 6 + c] = a;
  }
}

extern "C" void kernel_launch(void* const* d_in, const int* in_sizes, int n_in,
                              void* d_out, int out_size, void* d_ws, size_t ws_size,
                              hipStream_t stream) {
  (void)in_sizes; (void)n_in; (void)out_size; (void)ws_size;
  // setup_inputs order: pos, batch, cell, t_embedding, w1a,b1a,w2a,b2a,
  //                     wc1,bc1,wc2,bc2, wl1,bl1,wl2,bl2, wt,bt
  const int*   batch = (const int*)  d_in[1];
  const float* temb  = (const float*)d_in[3];
  const float* w1a   = (const float*)d_in[4];
  const float* b1a   = (const float*)d_in[5];
  const float* w2a   = (const float*)d_in[6];
  const float* b2a   = (const float*)d_in[7];
  const float* wc1   = (const float*)d_in[8];
  const float* bc1   = (const float*)d_in[9];
  const float* wc2   = (const float*)d_in[10];
  const float* bc2   = (const float*)d_in[11];
  const float* wl1   = (const float*)d_in[12];
  const float* bl1   = (const float*)d_in[13];
  const float* wl2   = (const float*)d_in[14];
  const float* bl2   = (const float*)d_in[15];
  const float* wt    = (const float*)d_in[16];
  const float* bt    = (const float*)d_in[17];

  // workspace: [W1T|W2T|WC1T|WTT] f16 (385024 B), then h_crystal f32 (4 MB)
  f16* W1T  = (f16*)d_ws;
  f16* W2T  = W1T + 256 * 256;
  f16* WC1T = W2T + 256 * 256;
  f16* WTT  = WC1T + 128 * 256;
  float* hc = (float*)((char*)d_ws + 385024);

  float* outCoord = (float*)d_out;                    // [N,3]
  float* outCell  = outCoord + (size_t)NN * 3;        // [B,6]
  float* outLogit = outCell + (size_t)BB * 6;         // [N,100]

  prep_wT<<<(256 * 256 + 255) / 256, 256, 0, stream>>>(w1a, W1T, 256, 256, 256);
  prep_wT<<<(256 * 256 + 255) / 256, 256, 0, stream>>>(w2a, W2T, 256, 256, 256);
  prep_wT<<<(128 * 256 + 255) / 256, 256, 0, stream>>>(wc1, WC1T, 256, 128, 128);
  prep_wT<<<(112 * 256 + 255) / 256, 256, 0, stream>>>(wt,  WTT,  256, 100, 112);
  hipMemsetAsync(hc, 0, (size_t)BB * HH * sizeof(float), stream);

  fused_atoms<<<(NN + TM - 1) / TM, 256, SMEM_FUSED, stream>>>(
      batch, temb, W1T, b1a, W2T, b2a, WC1T, bc1, wc2, bc2, WTT, bt,
      hc, outCoord, outLogit);

  const int smem_cell = (256 * 128 + 128 * 6 + 128 + 8 + CB * 256 + CB * 128) * 4;
  cell_head<<<BB / CB, 256, smem_cell, stream>>>(hc, wl1, bl1, wl2, bl2, outCell);
}
